// RelativeGlobalAttention_40793599377832
// MI455X (gfx1250) — matrix operand order repaired and checked by reading, not compiled, not run on hardware
//
#include <hip/hip_runtime.h>
#include <hip/hip_bf16.h>

typedef __bf16 bf16;
typedef __attribute__((ext_vector_type(16))) __bf16 v16bf;
typedef __attribute__((ext_vector_type(8)))  __bf16 v8bf;
typedef __attribute__((ext_vector_type(8)))  float  v8f;

#define DIMC  1024
#define HEADS 16
#define HD    64
#define NSEQ  1024

__device__ __forceinline__ v8f wmma_bf16(v16bf a, v16bf b, v8f c) {
    return __builtin_amdgcn_wmma_f32_16x16x32_bf16(false, a, false, b, (short)0, c, false, false);
}

__device__ __forceinline__ v8f zero8() {
    v8f z = {0.f,0.f,0.f,0.f,0.f,0.f,0.f,0.f};
    return z;
}

__device__ __forceinline__ v16bf load16(const bf16* p) {
    return *(const v16bf*)p;
}

__device__ __forceinline__ v16bf pair16(const bf16* plo, const bf16* phi) {
    v8bf lo = *(const v8bf*)plo;
    v8bf hi = *(const v8bf*)phi;
    v16bf a;
    #pragma unroll
    for (int i = 0; i < 8; ++i) { a[i] = lo[i]; a[8 + i] = hi[i]; }
    return a;
}

// ---------------------------------------------------------------------------
// Kernel 0: streaming fp32 -> bf16 conversion (one pass; frees the GEMM inner
// loops from all v_cvt VALU work and halves fragment bytes).
// ---------------------------------------------------------------------------
__global__ __launch_bounds__(256) void cvt_bf16_kernel(
    const float* __restrict__ src, bf16* __restrict__ dst, int n)
{
    const int i = (blockIdx.x * 256 + threadIdx.x) * 8;
    if (i >= n) return;
    v8bf v;
    #pragma unroll
    for (int k = 0; k < 8; ++k) v[k] = (bf16)src[i + k];
    *(v8bf*)(dst + i) = v;
}

// ---------------------------------------------------------------------------
// Kernel 1: qkv = x @ W_qkv^T (M=4096, N=3072, K=1024), all-bf16 operands.
// Wave tile 32 rows x 64 cols (2 A-frags share 4 B-frags -> 8 WMMA/k-step).
// Block = 8 waves -> 128 x 128 tile. Routes Q (pre-scaled), K, V-transposed.
// ---------------------------------------------------------------------------
__global__ __launch_bounds__(256) void qkv_kernel(
    const bf16* __restrict__ xb, const bf16* __restrict__ Wq,
    bf16* __restrict__ Qw, bf16* __restrict__ Kw, bf16* __restrict__ Vt)
{
    const int tid  = threadIdx.x;
    const int lane = tid & 31;
    const int w    = tid >> 5;
    const int col  = lane & 15;
    const int half = lane >> 4;

    const int r0  = blockIdx.x * 128 + (w >> 1) * 32;
    const int f0b = blockIdx.y * 128 + (w & 1) * 64;

    v8f acc[2][4];
    #pragma unroll
    for (int ra = 0; ra < 2; ++ra)
        #pragma unroll
        for (int t = 0; t < 4; ++t) acc[ra][t] = zero8();

    const bf16* xrow0 = xb + (size_t)(r0 + col) * DIMC;
    const bf16* xrow1 = xb + (size_t)(r0 + 16 + col) * DIMC;

    for (int kb = 0; kb < DIMC; kb += 32) {
        const int d0 = kb + half * 16;
        __builtin_prefetch(xrow0 + d0 + 64, 0, 3);
        v16bf a0 = pair16(xrow0 + d0, xrow0 + d0 + 8);
        v16bf a1 = pair16(xrow1 + d0, xrow1 + d0 + 8);
        #pragma unroll
        for (int t = 0; t < 4; ++t) {
            const bf16* wrow = Wq + (size_t)(f0b + t * 16 + col) * DIMC + kb + half * 16;
            v16bf bfr = load16(wrow);
            acc[0][t] = wmma_bf16(a0, bfr, acc[0][t]);
            acc[1][t] = wmma_bf16(a1, bfr, acc[1][t]);
        }
    }

    #pragma unroll
    for (int ra = 0; ra < 2; ++ra) {
        #pragma unroll
        for (int t = 0; t < 4; ++t) {
            const int f0 = f0b + t * 16;
            const int s  = f0 >> 10;               // 0=Q 1=K 2=V
            const int h  = (f0 >> 6) & 15;
            const int d  = (f0 & 63) + col;
            const int rowbase = r0 + ra * 16 + half * 8;
            const int b   = rowbase >> 10;
            const int nn0 = rowbase & 1023;
            const int bh  = b * HEADS + h;
            if (s == 2) {
                v8bf pack;
                #pragma unroll
                for (int r = 0; r < 8; ++r) pack[r] = (bf16)acc[ra][t][r];
                *(v8bf*)(Vt + ((size_t)bh * HD + d) * NSEQ + nn0) = pack;
            } else {
                bf16* dst = (s == 0) ? Qw : Kw;
                const float sc = (s == 0) ? 0.125f : 1.0f;   // fold 1/sqrt(64) into Q
                #pragma unroll
                for (int r = 0; r < 8; ++r)
                    dst[((size_t)bh * NSEQ + nn0 + r) * HD + d] = (bf16)(acc[ra][t][r] * sc);
            }
        }
    }
}

// ---------------------------------------------------------------------------
// Kernel 2: causal flash attention with relative-skew bias.
// rel[n,m] = q_n . Er[N-1-n+m] (m<=n): per 16x16 score tile the needed Er rows
// form a 31-wide band; compute band tiles R = q x Er^T with WMMA (rolling
// reuse, 2 new per 32-key chunk) and gather the diagonal via shuffles.
// ---------------------------------------------------------------------------
__global__ __launch_bounds__(256) void attn_kernel(
    const bf16* __restrict__ Qw, const bf16* __restrict__ Kw,
    const bf16* __restrict__ Vt, const bf16* __restrict__ Erb,
    bf16* __restrict__ AO)
{
    __shared__ bf16 Plds[8 * 16 * 32];   // per-wave 16x32 bf16 P staging (8 KB)

    const int tid  = threadIdx.x;
    const int lane = tid & 31;
    const int w    = tid >> 5;
    const int col  = lane & 15;
    const int half = lane >> 4;
    const int rowoff = half * 8;

    const int bh = blockIdx.y;
    const int h  = bh & 15;
    const int bq = bh >> 4;
    const int n0 = blockIdx.x * 128 + w * 16;

    bf16* pbase = Plds + w * 512;

    // Q fragments (scale pre-folded), two 16x32 A fragments covering d=0..63
    const bf16* qrow = Qw + ((size_t)bh * NSEQ + (n0 + col)) * HD;
    v16bf aq0 = pair16(qrow + half * 16,      qrow + half * 16 + 8);
    v16bf aq1 = pair16(qrow + half * 16 + 32, qrow + half * 16 + 40);

    // R tile: columns j = jbase..jbase+15 of q x Er^T (guard j >= N with zeros)
    auto relTile = [&](int jbase) -> v8f {
        v8f rt = zero8();
        const int j = jbase + col;
        #pragma unroll
        for (int ks = 0; ks < 2; ++ks) {
            v16bf eb;
            if (j < NSEQ) {
                eb = load16(Erb + (size_t)j * HD + ks * 32 + half * 16);
            } else {
                #pragma unroll
                for (int i = 0; i < 16; ++i) eb[i] = (bf16)0.0f;
            }
            rt = wmma_bf16(ks == 0 ? aq0 : aq1, eb, rt);
        }
        return rt;
    };

    const int jb = NSEQ - 16 - n0;       // >= 0 always
    v8f Ta = relTile(jb), Tb, Tc;

    v8f O[4] = { zero8(), zero8(), zero8(), zero8() };
    float mrow[8], lrow[8];
    #pragma unroll
    for (int r = 0; r < 8; ++r) { mrow[r] = -1e30f; lrow[r] = 0.0f; }

    for (int c0 = 0; c0 <= n0 + 15; c0 += 32) {
        Tb = relTile(jb + c0 + 16);
        Tc = relTile(jb + c0 + 32);

        // content: two 16x16 score tiles, K = head dim (two k=32 steps each)
        v8f S0 = zero8(), S1 = zero8();
        #pragma unroll
        for (int t = 0; t < 2; ++t) {
            const int m0 = c0 + t * 16;
            const bf16* krow = Kw + ((size_t)bh * NSEQ + (m0 + col)) * HD + half * 16;
            v16bf kb0 = load16(krow);
            v16bf kb1 = load16(krow + 32);
            v8f s = zero8();
            s = wmma_bf16(aq0, kb0, s);
            s = wmma_bf16(aq1, kb1, s);
            if (t == 0) S0 = s; else S1 = s;
        }

        // add skewed rel term via shuffle gather + causal mask
        #pragma unroll
        for (int r = 0; r < 8; ++r) {
            const int cj  = 15 - (r + rowoff) + col;         // 0..30
            const int src = (lane & 16) | (cj & 15);
            const float va = __shfl(Ta[r], src, 32);
            const float vb = __shfl(Tb[r], src, 32);
            const float vc = __shfl(Tc[r], src, 32);
            S0[r] += (cj < 16) ? va : vb;
            S1[r] += (cj < 16) ? vb : vc;
            const int row = n0 + r + rowoff;
            if (c0 + col      > row) S0[r] = -1e30f;
            if (c0 + 16 + col > row) S1[r] = -1e30f;
        }

        // online softmax, per-row stats across the 16-lane half groups
        float corr[8];
        #pragma unroll
        for (int r = 0; r < 8; ++r) {
            float mx = fmaxf(S0[r], S1[r]);
            #pragma unroll
            for (int off = 8; off >= 1; off >>= 1) mx = fmaxf(mx, __shfl_xor(mx, off, 32));
            const float mnew = fmaxf(mrow[r], mx);
            const float p0 = __expf(S0[r] - mnew);
            const float p1 = __expf(S1[r] - mnew);
            float sum = p0 + p1;
            #pragma unroll
            for (int off = 8; off >= 1; off >>= 1) sum += __shfl_xor(sum, off, 32);
            const float c = __expf(mrow[r] - mnew);
            lrow[r] = lrow[r] * c + sum;
            mrow[r] = mnew;
            corr[r] = c;
            pbase[(r + rowoff) * 32 + col]      = (bf16)p0;
            pbase[(r + rowoff) * 32 + 16 + col] = (bf16)p1;
        }
        #pragma unroll
        for (int t = 0; t < 4; ++t)
            #pragma unroll
            for (int r = 0; r < 8; ++r) O[t][r] *= corr[r];

        // P (16x32) A-fragment from LDS; V B-fragments from transposed Vt
        v16bf ap = pair16(pbase + col * 32 + half * 16,
                          pbase + col * 32 + 8 + half * 16);
        #pragma unroll
        for (int t = 0; t < 4; ++t) {
            const bf16* vrow = Vt + ((size_t)bh * HD + (t * 16 + col)) * NSEQ + c0 + half * 16;
            v16bf vfr = load16(vrow);
            O[t] = wmma_bf16(ap, vfr, O[t]);
        }

        Ta = Tc;
    }

    // normalize and write attention output [b][n][h*64+d] in bf16
    #pragma unroll
    for (int t = 0; t < 4; ++t) {
        const int c = h * HD + t * 16 + col;
        #pragma unroll
        for (int r = 0; r < 8; ++r) {
            const int n = n0 + r + rowoff;
            AO[((size_t)bq * NSEQ + n) * DIMC + c] = (bf16)(O[t][r] / lrow[r]);
        }
    }
}

// ---------------------------------------------------------------------------
// Kernel 3: out = AO @ W_proj^T + b_proj (M=4096, N=1024, K=1024), fp32 out.
// Same 32x64 wave tiling as kernel 1, all-bf16 operands.
// ---------------------------------------------------------------------------
__global__ __launch_bounds__(256) void proj_kernel(
    const bf16* __restrict__ AO, const bf16* __restrict__ Wp,
    const float* __restrict__ bias, float* __restrict__ out)
{
    const int tid  = threadIdx.x;
    const int lane = tid & 31;
    const int w    = tid >> 5;
    const int col  = lane & 15;
    const int half = lane >> 4;

    const int r0  = blockIdx.x * 128 + (w >> 1) * 32;
    const int o0b = blockIdx.y * 128 + (w & 1) * 64;

    v8f acc[2][4];
    #pragma unroll
    for (int ra = 0; ra < 2; ++ra)
        #pragma unroll
        for (int t = 0; t < 4; ++t) acc[ra][t] = zero8();

    const bf16* arow0 = AO + (size_t)(r0 + col) * DIMC;
    const bf16* arow1 = AO + (size_t)(r0 + 16 + col) * DIMC;

    for (int kb = 0; kb < DIMC; kb += 32) {
        const int d0 = kb + half * 16;
        __builtin_prefetch(arow0 + d0 + 64, 0, 3);
        v16bf a0 = pair16(arow0 + d0, arow0 + d0 + 8);
        v16bf a1 = pair16(arow1 + d0, arow1 + d0 + 8);
        #pragma unroll
        for (int t = 0; t < 4; ++t) {
            const bf16* wrow = Wp + (size_t)(o0b + t * 16 + col) * DIMC + kb + half * 16;
            v16bf bfr = load16(wrow);
            acc[0][t] = wmma_bf16(a0, bfr, acc[0][t]);
            acc[1][t] = wmma_bf16(a1, bfr, acc[1][t]);
        }
    }

    #pragma unroll
    for (int ra = 0; ra < 2; ++ra) {
        #pragma unroll
        for (int t = 0; t < 4; ++t) {
            const int o  = o0b + t * 16 + col;
            const float bo = bias[o];
            #pragma unroll
            for (int r = 0; r < 8; ++r)
                out[(size_t)(r0 + ra * 16 + r + half * 8) * DIMC + o] = acc[ra][t][r] + bo;
        }
    }
}

// ---------------------------------------------------------------------------
extern "C" void kernel_launch(void* const* d_in, const int* in_sizes, int n_in,
                              void* d_out, int out_size, void* d_ws, size_t ws_size,
                              hipStream_t stream) {
    const float* x     = (const float*)d_in[0];
    const float* Wqkv  = (const float*)d_in[1];
    const float* Wproj = (const float*)d_in[2];
    const float* bproj = (const float*)d_in[3];
    const float* Er    = (const float*)d_in[4];

    const size_t N_X    = (size_t)4096 * 1024;   // 4,194,304
    const size_t N_WQKV = (size_t)3072 * 1024;   // 3,145,728
    const size_t N_WP   = (size_t)1024 * 1024;   // 1,048,576
    const size_t N_ER   = (size_t)1024 * 64;     //    65,536
    const size_t N_HEAD = (size_t)4 * HEADS * NSEQ * HD;  // 4,194,304

    bf16* xb  = (bf16*)d_ws;
    bf16* Wqb = xb  + N_X;
    bf16* Wpb = Wqb + N_WQKV;
    bf16* Erb = Wpb + N_WP;
    bf16* Qw  = Erb + N_ER;
    bf16* Kw  = Qw  + N_HEAD;
    bf16* Vt  = Kw  + N_HEAD;
    bf16* AO  = Vt  + N_HEAD;

    cvt_bf16_kernel<<<(int)(N_X    / 2048), 256, 0, stream>>>(x,     xb,  (int)N_X);
    cvt_bf16_kernel<<<(int)(N_WQKV / 2048), 256, 0, stream>>>(Wqkv,  Wqb, (int)N_WQKV);
    cvt_bf16_kernel<<<(int)(N_WP   / 2048), 256, 0, stream>>>(Wproj, Wpb, (int)N_WP);
    cvt_bf16_kernel<<<(int)(N_ER   / 2048), 256, 0, stream>>>(Er,    Erb, (int)N_ER);

    qkv_kernel <<<dim3(32, 24), 256, 0, stream>>>(xb, Wqb, Qw, Kw, Vt);
    attn_kernel<<<dim3(8, 64),  256, 0, stream>>>(Qw, Kw, Vt, Erb, AO);
    proj_kernel<<<dim3(32, 8),  256, 0, stream>>>(AO, Wpb, bproj, (float*)d_out);
}
